// SingleHeadAttentionModel_88184268522037
// MI455X (gfx1250) — matrix operand order repaired and checked by reading, not compiled, not run on hardware
//
#include <hip/hip_runtime.h>

// ---------------------------------------------------------------------------
// SingleHeadAttention on MI455X (gfx1250): bf16 WMMA + TDM (tensor data mover)
// LDS-resident 64x1024 score panel; attn output streamed out via
// tensor_store_from_lds overlapping the attn@V WMMA phase.
// ---------------------------------------------------------------------------
#define B_SZ   128
#define S_LEN  1024
#define D_IN   49
#define D_K    64
#define N_CLS  1000

typedef __attribute__((ext_vector_type(16))) __bf16 v16bf;
typedef __attribute__((ext_vector_type(8)))  float  v8f;
typedef __attribute__((ext_vector_type(4)))  unsigned int v4u;
typedef __attribute__((ext_vector_type(8)))  int    v8i;
typedef __attribute__((ext_vector_type(4)))  int    v4i;

// Detect the TDM builtins (exist on both ROCm 7.2 clang-22 and clang-23
// amdgpu-toolchain, with different arity). Fall back to plain staging if absent.
#if defined(__has_builtin)
#  if __has_builtin(__builtin_amdgcn_tensor_load_to_lds) && \
      __has_builtin(__builtin_amdgcn_tensor_store_from_lds)
#    define HAVE_TDM 1
#  endif
#endif
#ifndef HAVE_TDM
#  define HAVE_TDM 0
#endif

#if HAVE_TDM
// Build a 1-D TDM descriptor: move n8 8-byte units between a contiguous
// global range and a contiguous LDS range (CDNA5 ISA).
__device__ inline void tdm_desc_1d(unsigned lds_off, unsigned long long gaddr,
                                   unsigned n8, v4u& g0, v8i& g1, v4i& g2, v4i& g3) {
    // Group 0: count=1 | lds_addr | global_addr[56:0] | type=2 ("image")
    g0[0] = 1u;
    g0[1] = lds_off;
    g0[2] = (unsigned)(gaddr & 0xFFFFFFFFull);
    g0[3] = (unsigned)((gaddr >> 32) & 0x01FFFFFFull) | (2u << 30);
    // Group 1: data_size=3 (8B); tensor_dim0 = tile_dim0 = n8; dim0 stride = n8
    g1[0] = (int)(3u << 16);                              // data_size = 8 bytes
    g1[1] = (int)((n8 & 0xFFFFu) << 16);                  // tensor_dim0[15:0]
    g1[2] = (int)(((n8 >> 16) & 0xFFFFu) | (1u << 16));   // tensor_dim0[31:16] | tensor_dim1=1
    g1[3] = (int)((n8 & 0xFFFFu) << 16);                  // tile_dim0 = n8
    g1[4] = 0;                                            // tile_dim1/2 unused
    g1[5] = (int)n8;                                      // tensor_dim0_stride[31:0]
    g1[6] = 0;
    g1[7] = 0;
    g2[0] = g2[1] = g2[2] = g2[3] = 0;
    g3[0] = g3[1] = g3[2] = g3[3] = 0;
}

__device__ inline void tdm_load_1d(unsigned lds_off, const void* gptr, unsigned n8) {
    v4u g0; v8i g1; v4i g2, g3;
    tdm_desc_1d(lds_off, (unsigned long long)(uintptr_t)gptr, n8, g0, g1, g2, g3);
#if __clang_major__ >= 23
    v8i g4; g4[0]=g4[1]=g4[2]=g4[3]=g4[4]=g4[5]=g4[6]=g4[7]=0;
    __builtin_amdgcn_tensor_load_to_lds(g0, g1, g2, g3, g4, 0);
#else
    __builtin_amdgcn_tensor_load_to_lds(g0, g1, g2, g3, 0);
#endif
}

__device__ inline void tdm_store_1d(unsigned lds_off, void* gptr, unsigned n8) {
    v4u g0; v8i g1; v4i g2, g3;
    tdm_desc_1d(lds_off, (unsigned long long)(uintptr_t)gptr, n8, g0, g1, g2, g3);
#if __clang_major__ >= 23
    v8i g4; g4[0]=g4[1]=g4[2]=g4[3]=g4[4]=g4[5]=g4[6]=g4[7]=0;
    __builtin_amdgcn_tensor_store_from_lds(g0, g1, g2, g3, g4, 0);
#else
    __builtin_amdgcn_tensor_store_from_lds(g0, g1, g2, g3, 0);
#endif
}
#endif // HAVE_TDM

// ---------------------------------------------------------------------------
// Kernel 0: zero the pooled accumulator region of the workspace
// ---------------------------------------------------------------------------
__global__ void zero_pooled_kernel(float* pooled, int n) {
    int i = blockIdx.x * blockDim.x + threadIdx.x;
    if (i < n) pooled[i] = 0.0f;
}

// ---------------------------------------------------------------------------
// Kernel 1: QKV projection.  x[B,S,49] (f32) @ W[49,64] (f32) -> bf16 Q/K/V.
// ---------------------------------------------------------------------------
__global__ __launch_bounds__(256)
void qkv_proj_kernel(const float* __restrict__ x,
                     const float* __restrict__ Wq,
                     const float* __restrict__ Wk,
                     const float* __restrict__ Wv,
                     __bf16* __restrict__ Qb,
                     __bf16* __restrict__ Kb,
                     __bf16* __restrict__ Vb) {
    __shared__ float xs[32 * D_IN];
    __shared__ float wqs[D_IN * D_K];
    __shared__ float wks[D_IN * D_K];
    __shared__ float wvs[D_IN * D_K];

    const int tid   = threadIdx.x;
    const int blk   = blockIdx.x;
    const int bIdx  = blk >> 5;
    const int s0    = (blk & 31) * 32;

    const size_t xbase = ((size_t)bIdx * S_LEN + s0) * D_IN;
    for (int i = tid; i < 32 * D_IN; i += 256) xs[i] = x[xbase + i];
    for (int i = tid; i < D_IN * D_K; i += 256) {
        wqs[i] = Wq[i]; wks[i] = Wk[i]; wvs[i] = Wv[i];
    }
    __syncthreads();

    for (int o = tid; o < 32 * D_K; o += 256) {
        const int r = o >> 6;
        const int k = o & 63;
        float aq = 0.f, ak = 0.f, av = 0.f;
        #pragma unroll 7
        for (int d = 0; d < D_IN; ++d) {
            const float xv = xs[r * D_IN + d];
            aq = fmaf(xv, wqs[d * D_K + k], aq);
            ak = fmaf(xv, wks[d * D_K + k], ak);
            av = fmaf(xv, wvs[d * D_K + k], av);
        }
        const size_t oi = ((size_t)bIdx * S_LEN + s0 + r) * D_K + k;
        Qb[oi] = (__bf16)aq;
        Kb[oi] = (__bf16)ak;
        Vb[oi] = (__bf16)av;
    }
}

// ---------------------------------------------------------------------------
// WMMA fragment loaders (layouts per CDNA5 ISA §7.12.2)
// ---------------------------------------------------------------------------
__device__ inline v16bf load_A_bf16(const __bf16* src, int rowBase, int kBase,
                                    int lane, int ld) {
    const int half = lane >> 4;
    const int m    = lane & 15;
    v16bf a;
    #pragma unroll
    for (int v = 0; v < 8; ++v) {
        const int kb = (v < 4) ? ((half ? 8 : 0) + 2 * v)
                               : ((half ? 24 : 16) + 2 * (v - 4));
        const __bf16* p = src + (rowBase + m) * ld + kBase + kb;
        a[2 * v]     = p[0];
        a[2 * v + 1] = p[1];
    }
    return a;
}

__device__ inline v16bf load_Bt_bf16(const __bf16* src, int nBase, int kBase,
                                     int lane, int ld) {
    const int half = lane >> 4;
    const int n    = lane & 15;
    v16bf b;
    #pragma unroll
    for (int v = 0; v < 8; ++v) {
        const int k = (v < 4) ? ((half ? 8 : 0) + 2 * v)
                              : ((half ? 24 : 16) + 2 * (v - 4));
        const __bf16* p = src + (nBase + n) * ld + kBase + k;
        b[2 * v]     = p[0];
        b[2 * v + 1] = p[1];
    }
    return b;
}

__device__ inline v16bf load_B_bf16(const __bf16* src, int nBase, int kBase,
                                    int lane, int ld) {
    const int half = lane >> 4;
    const int n    = lane & 15;
    v16bf b;
    #pragma unroll
    for (int v = 0; v < 8; ++v) {
        const int k = (v < 4) ? ((half ? 8 : 0) + 2 * v)
                              : ((half ? 24 : 16) + 2 * (v - 4));
        b[2 * v]     = src[(kBase + k) * ld + nBase + n];
        b[2 * v + 1] = src[(kBase + k + 1) * ld + nBase + n];
    }
    return b;
}

// ---------------------------------------------------------------------------
// Kernel 2: causal attention for one (batch, 64-query block).
//   LDS: sc[64][1024] f32 panel | qs/ks/vs 64x64 bf16 tiles (286720 B total)
// ---------------------------------------------------------------------------
#define SMEM_SC   0
#define SMEM_QS   262144
#define SMEM_KS   270336
#define SMEM_VS   278528
#define SMEM_TOT  286720

__global__ __launch_bounds__(256)
void attn_kernel(const __bf16* __restrict__ Qb,
                 const __bf16* __restrict__ Kb,
                 const __bf16* __restrict__ Vb,
                 float* __restrict__ attn_out,   // [B, S, S]
                 float* __restrict__ pooled) {   // [B, D_K] accumulated sums
    extern __shared__ char smem[];
    float*  sc = (float*)(smem + SMEM_SC);
    __bf16* qs = (__bf16*)(smem + SMEM_QS);
    __bf16* ks = (__bf16*)(smem + SMEM_KS);
    __bf16* vs = (__bf16*)(smem + SMEM_VS);

    const int tid  = threadIdx.x;
    const int lane = tid & 31;
    const int wave = tid >> 5;
    const int qb   = blockIdx.x;
    const int b    = blockIdx.y;
    const int q0   = qb * 64;
    const int nkb  = qb + 1;

    // --- stage Q tile (contiguous 4096 bf16 = 512 uint4) -------------------
    {
        const size_t base = ((size_t)b * S_LEN + q0) * D_K;
        const uint4* src = (const uint4*)(Qb + base);
        uint4* dst = (uint4*)qs;
        #pragma unroll
        for (int i = tid; i < 512; i += 256) dst[i] = src[i];
    }
    __syncthreads();

    // ======================= Phase 1: scores = Q K^T ========================
    for (int kb = 0; kb < nkb; ++kb) {
        const size_t kbase = ((size_t)b * S_LEN + kb * 64) * D_K;
#if HAVE_TDM
        if (wave == 0) {   // one TDM op stages the whole 8 KB K tile
            tdm_load_1d((unsigned)(uintptr_t)ks, (const void*)(Kb + kbase),
                        (64u * D_K * 2u) / 8u);
            __builtin_amdgcn_s_wait_tensorcnt(0);
        }
#else
        {
            const uint4* src = (const uint4*)(Kb + kbase);
            uint4* dst = (uint4*)ks;
            #pragma unroll
            for (int i = tid; i < 512; i += 256) dst[i] = src[i];
        }
#endif
        if (kb + 1 < nkb)
            __builtin_prefetch((const void*)(Kb + kbase + (size_t)64 * D_K), 0, 1);
        __syncthreads();

        for (int t = wave; t < 16; t += 8) {
            const int tm = t >> 2;
            const int tn = t & 3;
            v8f c = {};
            #pragma unroll
            for (int kc = 0; kc < D_K; kc += 32) {
                v16bf a  = load_A_bf16(qs, tm * 16, kc, lane, D_K);
                v16bf bm = load_Bt_bf16(ks, tn * 16, kc, lane, D_K);
                c = __builtin_amdgcn_wmma_f32_16x16x32_bf16(
                        false, a, false, bm, (short)0, c, false, false);
            }
            const int n0 = kb * 64 + tn * 16 + (lane & 15);
            const int mh = (lane >> 4) << 3;
            #pragma unroll
            for (int v = 0; v < 8; ++v) {
                const int m_loc = tm * 16 + v + mh;
                float val = c[v] * 0.125f;              // 1/sqrt(64)
                if (n0 > q0 + m_loc) val = -__builtin_inff();
                sc[m_loc * S_LEN + n0] = val;
            }
        }
        __syncthreads();
    }

    // ============ Phase 2: softmax; normalized probs kept in LDS ============
    for (int r = wave; r < 64; r += 8) {
        const int qg    = q0 + r;
        const int ncols = qg + 1;
        float* row = sc + r * S_LEN;

        float mx = -__builtin_inff();
        for (int c = lane; c < ncols; c += 32) mx = fmaxf(mx, row[c]);
        #pragma unroll
        for (int off = 16; off; off >>= 1) mx = fmaxf(mx, __shfl_xor(mx, off, 32));

        float sum = 0.f;
        for (int c = lane; c < ncols; c += 32) {
            const float p = __expf(row[c] - mx);
            row[c] = p;
            sum += p;
        }
        #pragma unroll
        for (int off = 16; off; off >>= 1) sum += __shfl_xor(sum, off, 32);
        const float inv = 1.0f / sum;

#if !HAVE_TDM
        float* orow = attn_out + ((size_t)b * S_LEN + qg) * S_LEN;
#endif
        for (int c = lane; c < S_LEN; c += 32) {
            const float p = (c < ncols) ? row[c] * inv : 0.f;
            row[c] = p;                 // full row (incl. causal zeros) in LDS
#if !HAVE_TDM
            orow[c] = p;
#endif
        }
    }
    __syncthreads();

#if HAVE_TDM
    // One TDM store streams the whole 256 KB attn panel LDS -> HBM, and it
    // overlaps with phase 3 below (which only *reads* sc).
    if (wave == 0) {
        float* gdst = attn_out + ((size_t)b * S_LEN + q0) * S_LEN;
        tdm_store_1d((unsigned)(uintptr_t)sc, (void*)gdst,
                     (64u * S_LEN * 4u) / 8u);   // 32768 x 8B
    }
#endif

    // ======================= Phase 3: hidden = attn @ V =====================
    v8f acc0 = {};
    v8f acc1 = {};
    for (int kb = 0; kb < nkb; ++kb) {
        const size_t vbase = ((size_t)b * S_LEN + kb * 64) * D_K;
#if HAVE_TDM
        if (wave == 0) {
            tdm_load_1d((unsigned)(uintptr_t)vs, (const void*)(Vb + vbase),
                        (64u * D_K * 2u) / 8u);
            __builtin_amdgcn_s_wait_tensorcnt(0);
        }
#else
        {
            const uint4* src = (const uint4*)(Vb + vbase);
            uint4* dst = (uint4*)vs;
            #pragma unroll
            for (int i = tid; i < 512; i += 256) dst[i] = src[i];
        }
#endif
        if (kb + 1 < nkb)
            __builtin_prefetch((const void*)(Vb + vbase + (size_t)64 * D_K), 0, 1);
        __syncthreads();

        #pragma unroll
        for (int i = 0; i < 2; ++i) {
            const int t  = wave + i * 8;
            const int tm = t >> 2;
            const int tn = t & 3;
            const int half = lane >> 4;
            const int m    = lane & 15;
            #pragma unroll
            for (int kc = 0; kc < 64; kc += 32) {
                v16bf a;
                const float* rowp = sc + (tm * 16 + m) * S_LEN + kb * 64 + kc;
                #pragma unroll
                for (int v = 0; v < 8; ++v) {
                    const int kpos = (v < 4) ? ((half ? 8 : 0) + 2 * v)
                                             : ((half ? 24 : 16) + 2 * (v - 4));
                    a[2 * v]     = (__bf16)rowp[kpos];
                    a[2 * v + 1] = (__bf16)rowp[kpos + 1];
                }
                v16bf bm = load_B_bf16(vs, tn * 16, kc, lane, D_K);
                if (i == 0)
                    acc0 = __builtin_amdgcn_wmma_f32_16x16x32_bf16(
                               false, a, false, bm, (short)0, acc0, false, false);
                else
                    acc1 = __builtin_amdgcn_wmma_f32_16x16x32_bf16(
                               false, a, false, bm, (short)0, acc1, false, false);
            }
        }
        __syncthreads();
    }

    // fold 64 query rows into pooled sums (fuses the mean over S)
    #pragma unroll
    for (int i = 0; i < 2; ++i) {
        const int t  = wave + i * 8;
        const int tn = t & 3;
        const v8f& a = (i == 0) ? acc0 : acc1;
        float s = 0.f;
        #pragma unroll
        for (int v = 0; v < 8; ++v) s += a[v];
        atomicAdd(pooled + b * D_K + tn * 16 + (lane & 15), s);
    }

#if HAVE_TDM
    // Drain the panel store before wave exit (S_ENDPGM also waits-idle).
    __builtin_amdgcn_s_wait_tensorcnt(0);
#endif
}

// ---------------------------------------------------------------------------
// Kernel 3: logits = (pooled / S) @ Wc + bc
// ---------------------------------------------------------------------------
__global__ __launch_bounds__(256)
void logits_kernel(const float* __restrict__ pooled,
                   const float* __restrict__ Wc,
                   const float* __restrict__ bc,
                   float* __restrict__ logits) {
    __shared__ float ps[D_K];
    const int b   = blockIdx.x;
    const int tid = threadIdx.x;
    if (tid < D_K) ps[tid] = pooled[b * D_K + tid] * (1.0f / (float)S_LEN);
    __syncthreads();
    for (int c = tid; c < N_CLS; c += 256) {
        float acc = bc[c];
        #pragma unroll
        for (int d = 0; d < D_K; ++d)
            acc = fmaf(ps[d], Wc[d * N_CLS + c], acc);
        logits[(size_t)b * N_CLS + c] = acc;
    }
}

// ---------------------------------------------------------------------------
// Launch
// ---------------------------------------------------------------------------
extern "C" void kernel_launch(void* const* d_in, const int* in_sizes, int n_in,
                              void* d_out, int out_size, void* d_ws, size_t ws_size,
                              hipStream_t stream) {
    const float* x  = (const float*)d_in[0];
    const float* Wq = (const float*)d_in[1];
    const float* Wk = (const float*)d_in[2];
    const float* Wv = (const float*)d_in[3];
    const float* Wc = (const float*)d_in[4];
    const float* bc = (const float*)d_in[5];

    float* logits   = (float*)d_out;
    float* attn_out = (float*)d_out + (size_t)B_SZ * N_CLS;

    float*  pooled = (float*)d_ws;
    __bf16* Qb = (__bf16*)((char*)d_ws + 32768);
    __bf16* Kb = Qb + (size_t)B_SZ * S_LEN * D_K;
    __bf16* Vb = Kb + (size_t)B_SZ * S_LEN * D_K;

    zero_pooled_kernel<<<(B_SZ * D_K + 255) / 256, 256, 0, stream>>>(pooled, B_SZ * D_K);
    qkv_proj_kernel<<<B_SZ * (S_LEN / 32), 256, 0, stream>>>(x, Wq, Wk, Wv, Qb, Kb, Vb);
    attn_kernel<<<dim3(S_LEN / 64, B_SZ), 256, SMEM_TOT, stream>>>(
        Qb, Kb, Vb, attn_out, pooled);
    logits_kernel<<<B_SZ, 256, 0, stream>>>(pooled, Wc, bc, logits);
}
